// surformer_2061584302218
// MI455X (gfx1250) — matrix operand order repaired and checked by reading, not compiled, not run on hardware
//
#include <hip/hip_runtime.h>
#include <math.h>

// ---------------------------------------------------------------------------
// Problem constants (from reference): B=1, C=64, H=W=128, NH=2, DH=32, K=7,
// HID = int(64*2.66) = 170, LRS = 0.1
// ---------------------------------------------------------------------------
#define NPIX   16384          // H*W
#define HH     128
#define WW     128
#define CCH    64
#define NFREQ  65             // rfft bins along W
#define FTOT   (CCH*HH*NFREQ) // 532480
#define NHID   170
#define NHID2  340
#define NHIDP  192            // K-padded gate width (multiple of 32)
#define NBIGP  384            // N-padded pin output width (multiple of 64)

typedef __attribute__((ext_vector_type(16))) __bf16        v16bf;
typedef __attribute__((ext_vector_type(8)))  float         v8f;
typedef __attribute__((ext_vector_type(4)))  unsigned int  u32x4;

static __device__ __forceinline__ unsigned short f2bf(float f) {
  unsigned int u = __float_as_uint(f);
  u += 0x7FFFu + ((u >> 16) & 1u);        // round-to-nearest-even
  return (unsigned short)(u >> 16);
}

union BfVec { v16bf v; u32x4 q[2]; };     // trivially-copyable fragment

// ---------------------------------------------------------------------------
// fp32 -> bf16 weight conversion / padding kernels
// ---------------------------------------------------------------------------
__global__ void cvt_f32_bf16(const float* __restrict__ in,
                             unsigned short* __restrict__ out, int n) {
  int i = blockIdx.x * 256 + threadIdx.x;
  if (i < n) out[i] = f2bf(in[i]);
}

// out (Ntot x K): rows >= Nsrc zero-filled
__global__ void cvt_pad_rows(const float* __restrict__ in,
                             unsigned short* __restrict__ out,
                             int Nsrc, int K, int Ntot) {
  int idx = blockIdx.x * 256 + threadIdx.x;
  if (idx >= Ntot * K) return;
  int nn = idx / K;
  out[idx] = (nn < Nsrc) ? f2bf(in[idx]) : (unsigned short)0;
}

// out (N x Kpad): cols >= Ksrc zero-filled
__global__ void cvt_pad_cols(const float* __restrict__ in,
                             unsigned short* __restrict__ out,
                             int N, int Ksrc, int Kpad) {
  int idx = blockIdx.x * 256 + threadIdx.x;
  if (idx >= N * Kpad) return;
  int kk = idx % Kpad, nn = idx / Kpad;
  out[idx] = (kk < Ksrc) ? f2bf(in[nn * Ksrc + kk]) : (unsigned short)0;
}

// Repack fs_w (Cout,Cin,3,3) -> bf16 [tap][cout][cin]
__global__ void repack_fsw(const float* __restrict__ w,
                           unsigned short* __restrict__ out) {
  int idx = blockIdx.x * 256 + threadIdx.x;
  if (idx >= 9 * 64 * 64) return;
  int ci = idx & 63, co = (idx >> 6) & 63, tap = idx >> 12;
  int ty = tap / 3, tx = tap % 3;
  out[idx] = f2bf(w[((co * 64 + ci) * 3 + ty) * 3 + tx]);
}

// ---------------------------------------------------------------------------
// WMMA GEMM:  C(MxN,f32) = A(MxK,bf16 row-major) * B^T   (B = weight NxK)
// All dims are exact tile multiples (padded buffers) -> NO guards.
// Per-wave 16x16 tile, K stepped by 32; block = 8 waves (2M x 4N) = 32x64.
// A fragment = two contiguous 8xbf16 runs (2x b128), B = the same two 8xbf16 runs.
// ---------------------------------------------------------------------------
__global__ __launch_bounds__(256)
void wmma_gemm_bf16(const unsigned short* __restrict__ A,
                    const unsigned short* __restrict__ B,
                    const float* __restrict__ bias,
                    float* __restrict__ C32,
                    unsigned short* __restrict__ Cb,
                    int N, int K) {
  const int lane = threadIdx.x & 31;
  const int wave = threadIdx.x >> 5;
  const int hi = lane >> 4;           // lane half
  const int lm = lane & 15;
  const int m0 = blockIdx.x * 32 + (wave & 1) * 16;
  const int n0 = blockIdx.y * 64 + (wave >> 1) * 16;
  const int m = m0 + lm;              // A row this lane feeds
  const int n = n0 + lm;              // B column this lane feeds
  const unsigned short* Arow = A + (size_t)m * K + hi * 8;   // chunk1 base
  const unsigned short* Brow = B + (size_t)n * K + hi * 8;
  v8f acc = {0.f, 0.f, 0.f, 0.f, 0.f, 0.f, 0.f, 0.f};
  for (int k0 = 0; k0 < K; k0 += 32) {
    __builtin_prefetch(Arow + k0 + 64, 0, 1);
    BfVec a, b;
    const u32x4* ap = (const u32x4*)(Arow + k0);
    a.q[0] = ap[0];                   // K = k0+hi*8   .. +7   (t=0..7)
    a.q[1] = ap[2];                   // K = k0+16+hi*8 .. +7  (t=8..15)
    const u32x4* bp = (const u32x4*)(Brow + k0);
    b.q[0] = bp[0];                   // K = k0+hi*8 .. +7, k0+16+hi*8 .. +7
    b.q[1] = bp[2];
    acc = __builtin_amdgcn_wmma_f32_16x16x32_bf16(false, a.v, false, b.v,
                                                  (short)0, acc, false, false);
  }
  const float bn = bias ? bias[n] : 0.f;
#pragma unroll
  for (int r = 0; r < 8; ++r) {
    int mm = m0 + hi * 8 + r;         // D: VGPR r -> M = r + 8*hi
    float v = acc[r] + bn;
    if (C32) C32[(size_t)mm * N + n] = v;
    if (Cb)  Cb[(size_t)mm * N + n] = f2bf(v);
  }
}

// ---------------------------------------------------------------------------
// 3x3 conv C->C as 9-tap WMMA GEMM over NHWC bf16 input (zero-pad borders).
// Wb layout: [tap][cout][cin] bf16.  Output yc (NPIX x 64) f32 (bias later).
// Border handling: one uniform branch per tap around the vector loads.
// ---------------------------------------------------------------------------
__global__ __launch_bounds__(256)
void wmma_conv3x3_bf16(const unsigned short* __restrict__ Xb,
                       const unsigned short* __restrict__ Wb,
                       float* __restrict__ C32) {
  const int lane = threadIdx.x & 31;
  const int wave = threadIdx.x >> 5;
  const int hi = lane >> 4;
  const int lm = lane & 15;
  const int m0 = blockIdx.x * 32 + (wave & 1) * 16;
  const int n0 = (wave >> 1) * 16;           // N = 64 covered by 4 waves
  const int m = m0 + lm;
  const int n = n0 + lm;
  const int pi = m >> 7, pj = m & 127;
  v8f acc = {0.f, 0.f, 0.f, 0.f, 0.f, 0.f, 0.f, 0.f};
  for (int tap = 0; tap < 9; ++tap) {
    const int dy = tap / 3 - 1, dx = tap % 3 - 1;
    const int ii = pi + dy, jj = pj + dx;
    const bool ok = (ii >= 0) && (ii < HH) && (jj >= 0) && (jj < WW);
    const unsigned short* arow = Xb + (size_t)((ii << 7) + jj) * 64 + hi * 8;
    const unsigned short* brow = Wb + (size_t)((tap << 6) + n) * 64 + hi * 8;
#pragma unroll
    for (int k0 = 0; k0 < 64; k0 += 32) {
      BfVec a, b;
      if (ok) {
        const u32x4* ap = (const u32x4*)(arow + k0);
        a.q[0] = ap[0];
        a.q[1] = ap[2];
      } else {
        a.q[0] = (u32x4)0u;
        a.q[1] = (u32x4)0u;
      }
      const u32x4* bp = (const u32x4*)(brow + k0);
      b.q[0] = bp[0];
      b.q[1] = bp[2];
      acc = __builtin_amdgcn_wmma_f32_16x16x32_bf16(false, a.v, false, b.v,
                                                    (short)0, acc, false, false);
    }
  }
#pragma unroll
  for (int r = 0; r < 8; ++r) {
    int mm = m0 + hi * 8 + r;
    C32[(size_t)mm * 64 + n] = acc[r];
  }
}

// ---------------------------------------------------------------------------
// LayerNorm over channels (NCHW): per-pixel mean / rsqrt(var)
// ---------------------------------------------------------------------------
__global__ void ln_stats(const float* __restrict__ x,
                         float* __restrict__ mu, float* __restrict__ rs) {
  int p = blockIdx.x * 256 + threadIdx.x;
  if (p >= NPIX) return;
  float s = 0.f, ss = 0.f;
#pragma unroll
  for (int c = 0; c < CCH; ++c) {
    float v = x[((size_t)c << 14) + p];
    s += v; ss += v * v;
  }
  float mn = s * (1.f / 64.f);
  float var = ss * (1.f / 64.f) - mn * mn;
  mu[p] = mn;
  rs[p] = rsqrtf(var + 1e-5f);
}

__global__ void ln_apply_nhwc_bf16(const float* __restrict__ x,
                                   const float* __restrict__ mu,
                                   const float* __restrict__ rs,
                                   const float* __restrict__ w,
                                   const float* __restrict__ b,
                                   unsigned short* __restrict__ out) {
  int idx = blockIdx.x * 256 + threadIdx.x;
  if (idx >= NPIX * CCH) return;
  int p = idx >> 6, c = idx & 63;
  out[idx] = f2bf((x[((size_t)c << 14) + p] - mu[p]) * rs[p] * w[c] + b[c]);
}

__global__ void ln_apply_nchw(const float* __restrict__ x,
                              const float* __restrict__ mu,
                              const float* __restrict__ rs,
                              const float* __restrict__ w,
                              const float* __restrict__ b,
                              float* __restrict__ out) {
  int idx = blockIdx.x * 256 + threadIdx.x;
  if (idx >= NPIX * CCH) return;
  int c = idx >> 14, p = idx & 16383;
  out[idx] = (x[idx] - mu[p]) * rs[p] * w[c] + b[c];
}

// dst(NCHW) = base(NCHW) + y(NHWC) [+ bias_c]
__global__ void add_nhwc(const float* __restrict__ base,
                         const float* __restrict__ y,
                         const float* __restrict__ bias,
                         float* __restrict__ dst) {
  int idx = blockIdx.x * 256 + threadIdx.x;
  if (idx >= NPIX * CCH) return;
  int c = idx >> 14, p = idx & 16383;
  dst[idx] = base[idx] + y[(size_t)p * 64 + c] + (bias ? bias[c] : 0.f);
}

// ---------------------------------------------------------------------------
// Neighborhood attention core: per (pixel, head) 7x7 window, dh=32.
// qkv layout: [p][s(3)][nh(2)][dh(32)] f32.  attn: [nh][p][49] f32.
// ---------------------------------------------------------------------------
__global__ __launch_bounds__(256)
void natten_kernel(const float* __restrict__ qkv,
                   const float* __restrict__ rpb,
                   float* __restrict__ attn,
                   unsigned short* __restrict__ aob,
                   int has_prev) {
  int idx = blockIdx.x * 256 + threadIdx.x;
  if (idx >= NPIX * 2) return;
  const int p = idx & (NPIX - 1);
  const int nh = idx >> 14;
  const int i = p >> 7, j = p & 127;
  int is = i - 3; is = is < 0 ? 0 : (is > 121 ? 121 : is);
  int js = j - 3; js = js < 0 ? 0 : (js > 121 ? 121 : js);

  float q[32];
  const float* qp = qkv + (size_t)p * 192 + nh * 32;
#pragma unroll
  for (int d = 0; d < 32; ++d) q[d] = qp[d] * 0.17677669529663687f;

  float lg[49];
#pragma unroll
  for (int a = 0; a < 7; ++a) {
    const int ih = is + a, rh = ih - i + 6;
#pragma unroll
    for (int bb = 0; bb < 7; ++bb) {
      const int iw = js + bb, rw = iw - j + 6;
      const float* kp = qkv + (size_t)((ih << 7) + iw) * 192 + 64 + nh * 32;
      float s = 0.f;
#pragma unroll
      for (int d = 0; d < 32; ++d) s += q[d] * kp[d];
      lg[a * 7 + bb] = s + rpb[nh * 169 + rh * 13 + rw];
    }
  }
  const size_t abase = ((size_t)nh * NPIX + p) * 49;
  if (has_prev) {
#pragma unroll
    for (int t = 0; t < 49; ++t) lg[t] += attn[abase + t];
  }
  float mx = -3.0e38f;
#pragma unroll
  for (int t = 0; t < 49; ++t) mx = fmaxf(mx, lg[t]);
  float sum = 0.f;
#pragma unroll
  for (int t = 0; t < 49; ++t) { lg[t] = __expf(lg[t] - mx); sum += lg[t]; }
  const float inv = 1.f / sum;
#pragma unroll
  for (int t = 0; t < 49; ++t) { lg[t] *= inv; attn[abase + t] = lg[t]; }

  float o[32];
#pragma unroll
  for (int d = 0; d < 32; ++d) o[d] = 0.f;
#pragma unroll
  for (int a = 0; a < 7; ++a) {
    const int ih = is + a;
#pragma unroll
    for (int bb = 0; bb < 7; ++bb) {
      const int iw = js + bb;
      const float w = lg[a * 7 + bb];
      const float* vp = qkv + (size_t)((ih << 7) + iw) * 192 + 128 + nh * 32;
#pragma unroll
      for (int d = 0; d < 32; ++d) o[d] += w * vp[d];
    }
  }
  unsigned short* op = aob + (size_t)p * 64 + nh * 32;
#pragma unroll
  for (int d = 0; d < 32; ++d) op[d] = f2bf(o[d]);
}

// ---------------------------------------------------------------------------
// Spectral branch: 128-pt DFTs with LDS twiddle tables.
// Freq layout: [c][h(128)][kw(65)]
// ---------------------------------------------------------------------------
__global__ __launch_bounds__(256)
void rfft_rows(const float* __restrict__ x, float* __restrict__ Fr,
               float* __restrict__ Fi) {
  __shared__ float twc[128], tws[128];
  if (threadIdx.x < 128) {
    float th = 6.283185307179586f * (float)threadIdx.x * (1.f / 128.f);
    twc[threadIdx.x] = cosf(th); tws[threadIdx.x] = sinf(th);
  }
  __syncthreads();
  int idx = blockIdx.x * 256 + threadIdx.x;
  if (idx >= FTOT) return;
  int kw = idx % 65, row = (idx / 65) & 127, c = idx / (65 * 128);
  const float* p = x + ((size_t)c << 14) + (row << 7);
  float re = 0.f, im = 0.f;
  for (int n = 0; n < 128; ++n) {
    int a = (kw * n) & 127;
    float v = p[n];
    re += v * twc[a]; im -= v * tws[a];
  }
  Fr[idx] = re; Fi[idx] = im;
}

__global__ __launch_bounds__(256)
void fft_cols(const float* __restrict__ Ar, const float* __restrict__ Ai,
              float* __restrict__ Br, float* __restrict__ Bi, int inverse) {
  __shared__ float twc[128], tws[128];
  if (threadIdx.x < 128) {
    float th = 6.283185307179586f * (float)threadIdx.x * (1.f / 128.f);
    twc[threadIdx.x] = cosf(th); tws[threadIdx.x] = sinf(th);
  }
  __syncthreads();
  int idx = blockIdx.x * 256 + threadIdx.x;
  if (idx >= FTOT) return;
  int kw = idx % 65, kh = (idx / 65) & 127, c = idx / (65 * 128);
  const float* pr = Ar + (size_t)c * (128 * 65) + kw;
  const float* pi = Ai + (size_t)c * (128 * 65) + kw;
  float re = 0.f, im = 0.f;
  for (int r = 0; r < 128; ++r) {
    int a = (kh * r) & 127;
    float xr = pr[r * 65], xi = pi[r * 65];
    if (inverse) { re += xr * twc[a] - xi * tws[a]; im += xi * twc[a] + xr * tws[a]; }
    else         { re += xr * twc[a] + xi * tws[a]; im += xi * twc[a] - xr * tws[a]; }
  }
  if (inverse) { re *= (1.f / 128.f); im *= (1.f / 128.f); }
  Br[idx] = re; Bi[idx] = im;
}

// depthwise 3x3 on 128x65 freq grid, SAME zero pad; optional leaky-relu(0.1)
__global__ void dw3x3_freq(const float* __restrict__ in,
                           const float* __restrict__ w,
                           const float* __restrict__ b,
                           float* __restrict__ out, int lrelu) {
  int idx = blockIdx.x * 256 + threadIdx.x;
  if (idx >= FTOT) return;
  int kw = idx % 65, r = (idx / 65) & 127, c = idx / (65 * 128);
  const float* base = in + (size_t)c * (128 * 65);
  float s = 0.f;
#pragma unroll
  for (int dy = -1; dy <= 1; ++dy) {
    int rr = r + dy;
    if (rr < 0 || rr >= 128) continue;
#pragma unroll
    for (int dx = -1; dx <= 1; ++dx) {
      int cc = kw + dx;
      if (cc < 0 || cc >= 65) continue;
      s += base[rr * 65 + cc] * w[c * 9 + (dy + 1) * 3 + (dx + 1)];
    }
  }
  s += b[c];
  if (lrelu) s = (s > 0.f) ? s : 0.1f * s;
  out[idx] = s;
}

// inverse rfft along W (65 bins -> 128 reals); writes NHWC bf16 for the conv
__global__ __launch_bounds__(256)
void irfft_rows(const float* __restrict__ Fr, const float* __restrict__ Fi,
                unsigned short* __restrict__ xsb) {
  __shared__ float twc[128], tws[128];
  if (threadIdx.x < 128) {
    float th = 6.283185307179586f * (float)threadIdx.x * (1.f / 128.f);
    twc[threadIdx.x] = cosf(th); tws[threadIdx.x] = sinf(th);
  }
  __syncthreads();
  int idx = blockIdx.x * 256 + threadIdx.x;
  if (idx >= NPIX * CCH) return;
  int n = idx & 127, r = (idx >> 7) & 127, c = idx >> 14;
  const float* pr = Fr + (size_t)c * (128 * 65) + r * 65;
  const float* pi = Fi + (size_t)c * (128 * 65) + r * 65;
  float s = pr[0] + ((n & 1) ? -pr[64] : pr[64]);
  for (int k = 1; k < 64; ++k) {
    int a = (k * n) & 127;
    s += 2.f * (pr[k] * twc[a] - pi[k] * tws[a]);
  }
  s *= (1.f / 128.f);
  xsb[(size_t)((r << 7) + n) * 64 + c] = f2bf(s);
}

// ---------------------------------------------------------------------------
// MLP: depthwise 3x3 on 340-ch NHWC(stride 384) + GELU gate
//  -> gb (NPIX x 192) bf16, channels >= 170 zeroed (K-padding for pout GEMM)
// ---------------------------------------------------------------------------
__global__ void dw_gate(const float* __restrict__ y,
                        const float* __restrict__ dwf,
                        unsigned short* __restrict__ gb) {
  int idx = blockIdx.x * 256 + threadIdx.x;
  if (idx >= NPIX * NHIDP) return;
  int c = idx % NHIDP, p = idx / NHIDP;
  if (c >= NHID) { gb[idx] = 0; return; }
  int i = p >> 7, j = p & 127;
  float v1 = 0.f, v2 = 0.f;
#pragma unroll
  for (int dy = -1; dy <= 1; ++dy) {
    int ii = i + dy;
    if (ii < 0 || ii >= HH) continue;
#pragma unroll
    for (int dx = -1; dx <= 1; ++dx) {
      int jj = j + dx;
      if (jj < 0 || jj >= WW) continue;
      const float* yp = y + (size_t)((ii << 7) + jj) * NBIGP;
      int t = (dy + 1) * 3 + (dx + 1);
      v1 += yp[c] * dwf[c * 9 + t];
      v2 += yp[c + NHID] * dwf[(c + NHID) * 9 + t];
    }
  }
  float g = v1 * 0.5f * (1.f + erff(v1 * 0.70710678118654752f)) * v2;
  gb[idx] = f2bf(g);
}

// ---------------------------------------------------------------------------
// ln4 + global average pool: gap[c] = w4[c]*mean_p((x-mu)*rs) + b4[c]
// ---------------------------------------------------------------------------
__global__ void gap_reduce(const float* __restrict__ xm,
                           const float* __restrict__ mu,
                           const float* __restrict__ rs,
                           const float* __restrict__ w4,
                           const float* __restrict__ b4,
                           float* __restrict__ gap) {
  __shared__ float sm[256];
  int c = blockIdx.x;
  float s = 0.f;
  for (int p = threadIdx.x; p < NPIX; p += 256)
    s += (xm[((size_t)c << 14) + p] - mu[p]) * rs[p];
  sm[threadIdx.x] = s;
  __syncthreads();
  for (int st = 128; st > 0; st >>= 1) {
    if (threadIdx.x < st) sm[threadIdx.x] += sm[threadIdx.x + st];
    __syncthreads();
  }
  if (threadIdx.x == 0) gap[c] = w4[c] * (sm[0] * (1.f / NPIX)) + b4[c];
}

// 64-pt channel FFT -> relu(scale) -> ifft real part
__global__ void cfm_kernel(const float* __restrict__ gap,
                           const float* __restrict__ cw,
                           const float* __restrict__ cb,
                           float* __restrict__ rec) {
  __shared__ float g[64], cr[64], ci[64];
  int t = threadIdx.x;
  g[t] = gap[t];
  __syncthreads();
  float fr = 0.f, fi = 0.f;
  for (int n = 0; n < 64; ++n) {
    float th = 6.283185307179586f * (float)((t * n) & 63) * (1.f / 64.f);
    fr += g[n] * cosf(th);
    fi -= g[n] * sinf(th);
  }
  float a = fr * cw[t] + cb[t];
  float b = fi * cw[t] + cb[t];
  cr[t] = a > 0.f ? a : 0.f;
  ci[t] = b > 0.f ? b : 0.f;
  __syncthreads();
  float s = 0.f;
  for (int k = 0; k < 64; ++k) {
    float th = 6.283185307179586f * (float)((k * t) & 63) * (1.f / 64.f);
    s += cr[k] * cosf(th) - ci[k] * sinf(th);
  }
  rec[t] = s * (1.f / 64.f);
}

__global__ void final_add(const float* __restrict__ xm,
                          const float* __restrict__ rec,
                          float* __restrict__ out) {
  int idx = blockIdx.x * 256 + threadIdx.x;
  if (idx >= NPIX * CCH) return;
  out[idx] = xm[idx] + rec[idx >> 14];
}

// ---------------------------------------------------------------------------
// Host launcher
// ---------------------------------------------------------------------------
extern "C" void kernel_launch(void* const* d_in, const int* in_sizes, int n_in,
                              void* d_out, int out_size, void* d_ws, size_t ws_size,
                              hipStream_t stream) {
  (void)in_sizes; (void)n_in; (void)out_size; (void)ws_size;
  // Inputs in setup_inputs() dict order: x, then params in insertion order.
  const float* x = (const float*)d_in[0];
  const float *lnw[4], *lnb[4];
  for (int i = 0; i < 4; ++i) { lnw[i] = (const float*)d_in[1 + 2 * i];
                                lnb[i] = (const float*)d_in[2 + 2 * i]; }
  const float *qkvw[3], *qkvbias[3], *rpb[3], *projw[3], *projbias[3];
  for (int s = 0; s < 3; ++s) {
    qkvw[s]     = (const float*)d_in[9 + 5 * s];
    qkvbias[s]  = (const float*)d_in[10 + 5 * s];
    rpb[s]      = (const float*)d_in[11 + 5 * s];
    projw[s]    = (const float*)d_in[12 + 5 * s];
    projbias[s] = (const float*)d_in[13 + 5 * s];
  }
  const float* fr_w1 = (const float*)d_in[24];
  const float* fr_b1 = (const float*)d_in[25];
  const float* fr_w2 = (const float*)d_in[26];
  const float* fr_b2 = (const float*)d_in[27];
  const float* fi_w1 = (const float*)d_in[28];
  const float* fi_b1 = (const float*)d_in[29];
  const float* fi_w2 = (const float*)d_in[30];
  const float* fi_b2 = (const float*)d_in[31];
  const float* fs_w  = (const float*)d_in[32];
  const float* fs_b  = (const float*)d_in[33];
  const float* pin_w = (const float*)d_in[34];
  const float* dwf_w = (const float*)d_in[35];
  const float* pout_w= (const float*)d_in[36];
  const float* cfm_w = (const float*)d_in[37];
  const float* cfm_b = (const float*)d_in[38];

  // --- workspace carve (~90 MB; everything stays L2-resident on MI455X) ---
  char* wsp = (char*)d_ws;
  size_t off = 0;
  auto alloc = [&](size_t bytes) -> void* {
    void* p = wsp + off;
    off += (bytes + 255) & ~(size_t)255;
    return p;
  };
  unsigned short* hb   = (unsigned short*)alloc((size_t)NPIX * 64 * 2);
  unsigned short* aob  = (unsigned short*)alloc((size_t)NPIX * 64 * 2);
  float* h32   = (float*)alloc((size_t)NPIX * 64 * 4);
  float* qkvf  = (float*)alloc((size_t)NPIX * 192 * 4);
  float* attnb = (float*)alloc((size_t)2 * NPIX * 49 * 4);
  float* x2    = (float*)alloc((size_t)NPIX * 64 * 4);
  float* h2    = (float*)alloc((size_t)NPIX * 64 * 4);
  float* mu    = (float*)alloc((size_t)NPIX * 4);
  float* rs    = (float*)alloc((size_t)NPIX * 4);
  float* Are   = (float*)alloc((size_t)FTOT * 4);
  float* Aim   = (float*)alloc((size_t)FTOT * 4);
  float* Bre   = (float*)alloc((size_t)FTOT * 4);
  float* Bim   = (float*)alloc((size_t)FTOT * 4);
  unsigned short* xsb = (unsigned short*)alloc((size_t)NPIX * 64 * 2);
  float* yc    = (float*)alloc((size_t)NPIX * 64 * 4);
  float* ybig  = (float*)alloc((size_t)NPIX * NBIGP * 4);   // stride 384
  unsigned short* gb = (unsigned short*)alloc((size_t)NPIX * NHIDP * 2);
  unsigned short* wq[3], *wp[3];
  for (int s = 0; s < 3; ++s) {
    wq[s] = (unsigned short*)alloc(192 * 64 * 2);
    wp[s] = (unsigned short*)alloc(64 * 64 * 2);
  }
  unsigned short* wpin  = (unsigned short*)alloc((size_t)NBIGP * 64 * 2);  // 384x64
  unsigned short* wpout = (unsigned short*)alloc((size_t)64 * NHIDP * 2);  // 64x192
  unsigned short* fswb  = (unsigned short*)alloc(9 * 64 * 64 * 2);
  float* gapb = (float*)alloc(256);
  float* recb = (float*)alloc(256);

  auto nb = [](int n) { return (n + 255) / 256; };

  // --- weight conversion to bf16 (with tile padding) ---
  for (int s = 0; s < 3; ++s) {
    cvt_f32_bf16<<<nb(192 * 64), 256, 0, stream>>>(qkvw[s], wq[s], 192 * 64);
    cvt_f32_bf16<<<nb(64 * 64), 256, 0, stream>>>(projw[s], wp[s], 64 * 64);
  }
  cvt_pad_rows<<<nb(NBIGP * 64), 256, 0, stream>>>(pin_w, wpin, NHID2, 64, NBIGP);
  cvt_pad_cols<<<nb(64 * NHIDP), 256, 0, stream>>>(pout_w, wpout, 64, NHID, NHIDP);
  repack_fsw<<<nb(9 * 64 * 64), 256, 0, stream>>>(fs_w, fswb);

  // --- stage 1: ln1 + 3x neighborhood attention (WMMA GEMMs) ---
  ln_stats<<<nb(NPIX), 256, 0, stream>>>(x, mu, rs);
  ln_apply_nhwc_bf16<<<nb(NPIX * 64), 256, 0, stream>>>(x, mu, rs, lnw[0], lnb[0], hb);
  for (int s = 0; s < 3; ++s) {
    wmma_gemm_bf16<<<dim3(NPIX / 32, 3), 256, 0, stream>>>(
        hb, wq[s], qkvbias[s], qkvf, nullptr, 192, 64);
    natten_kernel<<<nb(NPIX * 2), 256, 0, stream>>>(qkvf, rpb[s], attnb, aob, s > 0);
    wmma_gemm_bf16<<<dim3(NPIX / 32, 1), 256, 0, stream>>>(
        aob, wp[s], projbias[s], h32, hb, 64, 64);
  }
  add_nhwc<<<nb(NPIX * 64), 256, 0, stream>>>(x, h32, nullptr, x2);

  // --- stage 2: spectral branch ---
  ln_stats<<<nb(NPIX), 256, 0, stream>>>(x2, mu, rs);
  ln_apply_nchw<<<nb(NPIX * 64), 256, 0, stream>>>(x2, mu, rs, lnw[1], lnb[1], h2);
  rfft_rows<<<nb(FTOT), 256, 0, stream>>>(h2, Are, Aim);
  fft_cols<<<nb(FTOT), 256, 0, stream>>>(Are, Aim, Bre, Bim, 0);
  dw3x3_freq<<<nb(FTOT), 256, 0, stream>>>(Bre, fr_w1, fr_b1, Are, 1);
  dw3x3_freq<<<nb(FTOT), 256, 0, stream>>>(Are, fr_w2, fr_b2, Bre, 0);
  dw3x3_freq<<<nb(FTOT), 256, 0, stream>>>(Bim, fi_w1, fi_b1, Aim, 1);
  dw3x3_freq<<<nb(FTOT), 256, 0, stream>>>(Aim, fi_w2, fi_b2, Bim, 0);
  fft_cols<<<nb(FTOT), 256, 0, stream>>>(Bre, Bim, Are, Aim, 1);
  irfft_rows<<<nb(NPIX * 64), 256, 0, stream>>>(Are, Aim, xsb);
  wmma_conv3x3_bf16<<<NPIX / 32, 256, 0, stream>>>(xsb, fswb, yc);
  add_nhwc<<<nb(NPIX * 64), 256, 0, stream>>>(x2, yc, fs_b, x2);

  // --- stage 3: gated MLP ---
  ln_stats<<<nb(NPIX), 256, 0, stream>>>(x2, mu, rs);
  ln_apply_nhwc_bf16<<<nb(NPIX * 64), 256, 0, stream>>>(x2, mu, rs, lnw[2], lnb[2], hb);
  wmma_gemm_bf16<<<dim3(NPIX / 32, NBIGP / 64), 256, 0, stream>>>(
      hb, wpin, nullptr, ybig, nullptr, NBIGP, 64);
  dw_gate<<<nb(NPIX * NHIDP), 256, 0, stream>>>(ybig, dwf_w, gb);
  wmma_gemm_bf16<<<dim3(NPIX / 32, 1), 256, 0, stream>>>(
      gb, wpout, nullptr, yc, nullptr, 64, NHIDP);
  add_nhwc<<<nb(NPIX * 64), 256, 0, stream>>>(x2, yc, nullptr, x2);

  // --- stage 4: ln4 + GAP + channel-FFT modulation ---
  ln_stats<<<nb(NPIX), 256, 0, stream>>>(x2, mu, rs);
  gap_reduce<<<64, 256, 0, stream>>>(x2, mu, rs, lnw[3], lnb[3], gapb);
  cfm_kernel<<<1, 64, 0, stream>>>(gapb, cfm_w, cfm_b, recb);
  final_add<<<nb(NPIX * 64), 256, 0, stream>>>(x2, recb, (float*)d_out);
}
